// MultiHeadAttention_1889785610392
// MI455X (gfx1250) — compile-verified, matrix operand order checked by reading, not run on hardware
//
#include <hip/hip_runtime.h>
#include <hip/hip_bf16.h>
#include <stdint.h>

#define D_MODEL 1024
#define N_HEADS 16
#define DEPTH   64
#define SEQ     2048
#define BATCH   2

typedef __attribute__((ext_vector_type(16))) __bf16 v16bf;
typedef __attribute__((ext_vector_type(8)))  float  v8f;
typedef __attribute__((ext_vector_type(4)))  unsigned int v4u;
typedef __attribute__((ext_vector_type(4)))  int    v4i;
typedef __attribute__((ext_vector_type(8)))  int    v8i;

#if __has_builtin(__builtin_amdgcn_tensor_load_to_lds) && \
    __has_builtin(__builtin_amdgcn_s_wait_tensorcnt)
#define HAVE_TDM 1
#else
#define HAVE_TDM 0
#endif

// Pack two f32 -> two bf16 (round-half-up) in one v_perm_b32.
__device__ __forceinline__ unsigned pkbf(float a, float b) {
    union { float f; unsigned u; } ca, cb; ca.f = a; cb.f = b;
    return __builtin_amdgcn_perm(cb.u + 0x8000u, ca.u + 0x8000u, 0x07060302u);
}

__device__ __forceinline__ unsigned short f2bf(float f) {
    union { float f; unsigned u; } c; c.f = f;
    return (unsigned short)((c.u + 0x8000u) >> 16);
}

// 16x32 bf16 WMMA A/B fragment from a bf16 (u16) buffer: two 16B loads.
// lane<16 holds K = {0..7, 16..23}; lane>=16 holds K = {8..15, 24..31}.
__device__ __forceinline__ v16bf ldfrag_bf16(const unsigned short* __restrict__ base,
                                             size_t off) {
    union { v16bf v; uint4 q[2]; } f;
    f.q[0] = *(const uint4*)(base + off);
    f.q[1] = *(const uint4*)(base + off + 16);
    return f.v;
}

#define WMMA_BF16(A, Bm, C) __builtin_amdgcn_wmma_f32_16x16x32_bf16( \
    false, (A), false, (Bm), (short)0, (C), false, false)

// -------------------- Kernel 0: one-shot f32 -> bf16 conversion --------------------
__global__ __launch_bounds__(256)
void cvt_bf16(const float* __restrict__ X,
              const float* __restrict__ Wq, const float* __restrict__ Wk,
              const float* __restrict__ Wv, const float* __restrict__ Wo,
              unsigned short* __restrict__ xb,
              unsigned short* __restrict__ wqb, unsigned short* __restrict__ wkb,
              unsigned short* __restrict__ wvb, unsigned short* __restrict__ wob) {
    const int chunk = blockIdx.y;
    const float* src;
    unsigned short* dst;
    if (chunk < 4) {
        src = X  + (size_t)chunk * (1u << 20);
        dst = xb + (size_t)chunk * (1u << 20);
    } else {
        const float* s4[4]    = {Wq, Wk, Wv, Wo};
        unsigned short* d4[4] = {wqb, wkb, wvb, wob};
        src = s4[chunk - 4];
        dst = d4[chunk - 4];
    }
    size_t i = ((size_t)blockIdx.x * 256 + threadIdx.x) * 8;
    float4 f0 = *(const float4*)(src + i);
    float4 f1 = *(const float4*)(src + i + 4);
    uint4 o;
    o.x = pkbf(f0.x, f0.y); o.y = pkbf(f0.z, f0.w);
    o.z = pkbf(f1.x, f1.y); o.w = pkbf(f1.z, f1.w);
    *(uint4*)(dst + i) = o;
}

// -------------------- Kernel 1: QKV projection + bias + RoPE (ping-pong WMMA) ----------
// grid = (4096/32, H, 3), block = 32. Wave computes 32 rows x one head (64 cols).
__global__ __launch_bounds__(32)
void qkv_rope_wmma(const unsigned short* __restrict__ Xb,
                   const unsigned short* __restrict__ Wqb,
                   const unsigned short* __restrict__ Wkb,
                   const unsigned short* __restrict__ Wvb,
                   const float* __restrict__ bq, const float* __restrict__ bk,
                   const float* __restrict__ bv,
                   unsigned short* __restrict__ qws,
                   unsigned short* __restrict__ kws,
                   unsigned short* __restrict__ vws) {
    const int z = blockIdx.z;
    const unsigned short* W = (z == 0) ? Wqb : (z == 1) ? Wkb : Wvb;
    const float* B          = (z == 0) ? bq  : (z == 1) ? bk  : bv;
    unsigned short* dst     = (z == 0) ? qws : (z == 1) ? kws : vws;

    const int lane = threadIdx.x & 31, half = lane >> 4, ln = lane & 15;
    const int mBase = blockIdx.x * 32;
    const int head  = blockIdx.y;
    const int nBase = head * DEPTH;

    const size_t aoff0 = (size_t)(mBase + ln)      * D_MODEL + half * 8;
    const size_t aoff1 = (size_t)(mBase + 16 + ln) * D_MODEL + half * 8;
    size_t boff[4];
#pragma unroll
    for (int j = 0; j < 4; j++)
        boff[j] = (size_t)(nBase + 16 * j + ln) * D_MODEL + half * 8;

    // explicit 2-stage ping-pong: no register rotation, loads overlap WMMAs
    v16bf a0A = ldfrag_bf16(Xb, aoff0);
    v16bf a1A = ldfrag_bf16(Xb, aoff1);
    v16bf bA[4], a0B, a1B, bB[4];
#pragma unroll
    for (int j = 0; j < 4; j++) bA[j] = ldfrag_bf16(W, boff[j]);

    v8f acc[2][4] = {};
    for (int kb = 0; kb < D_MODEL; kb += 64) {
        // stage B loads (kb+32 is always < D_MODEL)
        __builtin_prefetch(Xb + aoff0 + kb + 128, 0, 3);
        a0B = ldfrag_bf16(Xb, aoff0 + kb + 32);
        a1B = ldfrag_bf16(Xb, aoff1 + kb + 32);
#pragma unroll
        for (int j = 0; j < 4; j++) bB[j] = ldfrag_bf16(W, boff[j] + kb + 32);
        // compute stage A
#pragma unroll
        for (int j = 0; j < 4; j++) {
            acc[0][j] = WMMA_BF16(a0A, bA[j], acc[0][j]);
            acc[1][j] = WMMA_BF16(a1A, bA[j], acc[1][j]);
        }
        // stage A loads for kb+64
        if (kb + 64 < D_MODEL) {
            a0A = ldfrag_bf16(Xb, aoff0 + kb + 64);
            a1A = ldfrag_bf16(Xb, aoff1 + kb + 64);
#pragma unroll
            for (int j = 0; j < 4; j++) bA[j] = ldfrag_bf16(W, boff[j] + kb + 64);
        }
        // compute stage B
#pragma unroll
        for (int j = 0; j < 4; j++) {
            acc[0][j] = WMMA_BF16(a0B, bB[j], acc[0][j]);
            acc[1][j] = WMMA_BF16(a1B, bB[j], acc[1][j]);
        }
    }

#pragma unroll
    for (int t = 0; t < 2; t++) {
        const int mB    = mBase + 16 * t;
        const int sBase = mB % SEQ;   // 16-row tile never crosses a batch boundary
        const int bIdx  = mB / SEQ;

        float res[4][8];
#pragma unroll
        for (int j = 0; j < 4; j++) {
            float bb = B[nBase + 16 * j + ln];
#pragma unroll
            for (int r = 0; r < 8; r++) res[j][r] = acc[t][j][r] + bb;
        }

        if (z < 2) {  // RoPE: rot(d<32) = -x[d+32]; rot(d>=32) = x[d-32]
            float out[4][8];
#pragma unroll
            for (int j = 0; j < 4; j++) {
                int d32 = (16 * j + ln) & 31;
                float invf = __expf(-0.28782313662425576f * (float)d32);  // 10000^(-d32/32)
#pragma unroll
                for (int r = 0; r < 8; r++) {
                    float sp  = (float)(sBase + r + 8 * half);
                    float ang = sp * invf;
                    float cs = __cosf(ang), sn = __sinf(ang);
                    float rot = (j < 2) ? -res[j + 2][r] : res[j - 2][r];
                    out[j][r] = res[j][r] * cs + rot * sn;
                }
            }
#pragma unroll
            for (int j = 0; j < 4; j++)
#pragma unroll
                for (int r = 0; r < 8; r++) res[j][r] = out[j][r];
        }

#pragma unroll
        for (int j = 0; j < 4; j++) {
            int d = 16 * j + ln;
#pragma unroll
            for (int r = 0; r < 8; r++) {
                int s = sBase + r + 8 * half;
                dst[(((size_t)bIdx * N_HEADS + head) * SEQ + s) * DEPTH + d] = f2bf(res[j][r]);
            }
        }
    }
}

// -------------------- Kernel 2: flash attention (TDM K-staging + transposed logits) ----------
#define KROW 72   // LDS row stride (bf16) for K tile [32][64]: 32DW data + 4DW pad (TDM pad)
#define VROW 40   // LDS row stride for transposed V tile [64][32]

#if HAVE_TDM
// TENSOR_LOAD_TO_LDS of a 32x64 bf16 tile (row stride 64 elems) into LDS with
// 4-DWORD padding after every 32 DWORDs (-> KROW=72 row stride).
__device__ __forceinline__ void tdm_load_k_tile(const unsigned short* gsrc,
                                                unsigned lds_addr) {
    unsigned long long ga = (unsigned long long)(uintptr_t)gsrc;
    v4u g0;
    g0[0] = 1u;                                        // count=1, user descriptor
    g0[1] = lds_addr;                                  // LDS byte address
    g0[2] = (unsigned)(ga & 0xFFFFFFFFu);              // global_addr[31:0]
    g0[3] = (unsigned)((ga >> 32) & 0x1FFFFFFu) | 0x80000000u;  // addr[56:32] | type=2

    const unsigned td0 = 64, td1 = 1u << 20;           // tensor dims (no OOB)
    const unsigned tile0 = 64, tile1 = 32;
    const unsigned long long st0 = 64;                 // row stride in elems
    v8i g1;
    g1[0] = (int)((1u << 16)      // data_size = 2 bytes
                | (1u << 20)      // pad_enable
                | (4u << 22)      // pad_interval: 32 DWORDs
                | (3u << 25));    // pad_amount:   4 DWORDs
    g1[1] = (int)((td0 & 0xFFFFu) << 16);
    g1[2] = (int)(((td0 >> 16) & 0xFFFFu) | ((td1 & 0xFFFFu) << 16));
    g1[3] = (int)(((td1 >> 16) & 0xFFFFu) | (tile0 << 16));
    g1[4] = (int)(tile1);                              // tile_dim1 | tile_dim2=0
    g1[5] = (int)(st0 & 0xFFFFFFFFu);
    g1[6] = (int)((st0 >> 32) & 0xFFFFu);              // | dim1_stride[15:0]=0
    g1[7] = 0;

    v4i z4 = {0, 0, 0, 0};
#if __has_include(<hip/amd_detail/amd_gfx1250_TDM.h>)
    v8i z8 = {0, 0, 0, 0, 0, 0, 0, 0};                 // therock toolchain: 6-arg form
    __builtin_amdgcn_tensor_load_to_lds(g0, g1, z4, z4, z8, 0);
#else
    __builtin_amdgcn_tensor_load_to_lds(g0, g1, z4, z4, 0);  // ROCm 7.2: 5-arg form
#endif
}
#endif

// grid = (SEQ/128, H, B), block = 256 (8 waves, each owns 16 query rows).
__global__ __launch_bounds__(256)
void attn_flash_wmma(const unsigned short* __restrict__ qws,
                     const unsigned short* __restrict__ kws,
                     const unsigned short* __restrict__ vws,
                     unsigned short* __restrict__ ows) {
    __shared__ unsigned short Kl[32 * KROW];
    __shared__ unsigned short Vl[64 * VROW];

    const int tid = threadIdx.x, lane = tid & 31, half = lane >> 4, ln = lane & 15;
    const int wave = tid >> 5;
    const int h = blockIdx.y, b = blockIdx.z, bh = b * N_HEADS + h;
    const int qBase = blockIdx.x * 128 + wave * 16;

    const unsigned short* qp = qws + (size_t)bh * SEQ * DEPTH;
    const unsigned short* kp = kws + (size_t)bh * SEQ * DEPTH;
    const unsigned short* vp = vws + (size_t)bh * SEQ * DEPTH;

    // Q^T as B-fragments: lane = qrow (ln), K-dim = dh, contiguous bf16 loads.
    v16bf Qf[2];
#pragma unroll
    for (int dk = 0; dk < 2; dk++)
        Qf[dk] = ldfrag_bf16(qp, (size_t)(qBase + ln) * DEPTH + dk * 32 + half * 8);

    v8f O[4] = {};
    float mrow = -1.0e30f, lrow = 0.0f;

    for (int kt = 0; kt < SEQ / 32; ++kt) {
        const int keyBase = kt * 32;
        const int key = tid >> 3, dc = (tid & 7) * 8;
#if HAVE_TDM
        if (wave == 0)   // async K tile DMA: global -> LDS with pad (row stride 72)
            tdm_load_k_tile(kp + (size_t)keyBase * DEPTH,
                            (unsigned)(uintptr_t)&Kl[0]);
#else
        {
            uint4 kq = *(const uint4*)(kp + (size_t)(keyBase + key) * DEPTH + dc);
            *(uint4*)(&Kl[key * KROW + dc]) = kq;
        }
#endif
        {   // V staged transposed (TDM cannot transpose): [64][32] layout
            uint4 vq = *(const uint4*)(vp + (size_t)(keyBase + key) * DEPTH + dc);
            const unsigned short* vs = (const unsigned short*)&vq;
#pragma unroll
            for (int i = 0; i < 8; i++) Vl[(dc + i) * VROW + key] = vs[i];
        }
#if HAVE_TDM
        if (wave == 0) __builtin_amdgcn_s_wait_tensorcnt(0);
#endif
        __syncthreads();

        // logits^T tiles: Ct[t][r] = logit for key (r + 8*half + 16*t), qrow = ln
        v8f Ct[2] = {};
#pragma unroll
        for (int t = 0; t < 2; t++) {
#pragma unroll
            for (int dk = 0; dk < 2; dk++) {
                v16bf ka = ldfrag_bf16(Kl, (size_t)(t * 16 + ln) * KROW + dk * 32 + half * 8);
                Ct[t] = WMMA_BF16(ka, Qf[dk], Ct[t]);
            }
        }

        // online softmax over 32 keys
        float p0[8], p1[8], pm = -1.0e30f;
#pragma unroll
        for (int r = 0; r < 8; r++) {
            p0[r] = Ct[0][r] * 0.125f;
            p1[r] = Ct[1][r] * 0.125f;
            pm = fmaxf(pm, fmaxf(p0[r], p1[r]));
        }
        pm = fmaxf(pm, __shfl_xor(pm, 16, 32));
        float mnew = fmaxf(mrow, pm);
        float ps = 0.0f;
#pragma unroll
        for (int r = 0; r < 8; r++) {
            p0[r] = __expf(p0[r] - mnew);
            p1[r] = __expf(p1[r] - mnew);
            ps += p0[r] + p1[r];
        }
        ps += __shfl_xor(ps, 16, 32);
        float corr = __expf(mrow - mnew);
        lrow = lrow * corr + ps;
        mrow = mnew;

        float cr[8];
#pragma unroll
        for (int r = 0; r < 8; r++) cr[r] = __shfl(corr, half * 8 + r, 32);
#pragma unroll
        for (int j = 0; j < 4; j++)
#pragma unroll
            for (int r = 0; r < 8; r++) O[j][r] *= cr[r];

        // P^T C-layout == P A-fragment layout: local pack, 1 v_perm per pair
        union { v16bf v; unsigned w[8]; } P;
#pragma unroll
        for (int r = 0; r < 4; r++) {
            P.w[r]     = pkbf(p0[2 * r], p0[2 * r + 1]);
            P.w[4 + r] = pkbf(p1[2 * r], p1[2 * r + 1]);
        }

#pragma unroll
        for (int j = 0; j < 4; j++) {
            v16bf vb = ldfrag_bf16(Vl, (size_t)(16 * j + ln) * VROW + half * 8);
            O[j] = WMMA_BF16(P.v, vb, O[j]);
        }
        __syncthreads();
    }

    float linv = 1.0f / lrow;
    float lr[8];
#pragma unroll
    for (int r = 0; r < 8; r++) lr[r] = __shfl(linv, half * 8 + r, 32);
#pragma unroll
    for (int j = 0; j < 4; j++) {
        int d = 16 * j + ln;
#pragma unroll
        for (int r = 0; r < 8; r++) {
            int s = qBase + r + 8 * half;
            ows[((size_t)b * SEQ + s) * D_MODEL + h * DEPTH + d] = f2bf(O[j][r] * lr[r]);
        }
    }
}

// -------------------- Kernel 3: output projection + gamma (ping-pong WMMA) -------------
__global__ __launch_bounds__(32)
void oproj_wmma(const unsigned short* __restrict__ Xb,
                const unsigned short* __restrict__ Wob,
                const float* __restrict__ bo,
                const float* __restrict__ gamma, float* __restrict__ out) {
    const int lane = threadIdx.x & 31, half = lane >> 4, ln = lane & 15;
    const int mBase = blockIdx.x * 32;
    const int nBase = blockIdx.y * 64;

    const size_t aoff0 = (size_t)(mBase + ln)      * D_MODEL + half * 8;
    const size_t aoff1 = (size_t)(mBase + 16 + ln) * D_MODEL + half * 8;
    size_t boff[4];
#pragma unroll
    for (int j = 0; j < 4; j++)
        boff[j] = (size_t)(nBase + 16 * j + ln) * D_MODEL + half * 8;

    v16bf a0A = ldfrag_bf16(Xb, aoff0);
    v16bf a1A = ldfrag_bf16(Xb, aoff1);
    v16bf bA[4], a0B, a1B, bB[4];
#pragma unroll
    for (int j = 0; j < 4; j++) bA[j] = ldfrag_bf16(Wob, boff[j]);

    v8f acc[2][4] = {};
    for (int kb = 0; kb < D_MODEL; kb += 64) {
        a0B = ldfrag_bf16(Xb, aoff0 + kb + 32);
        a1B = ldfrag_bf16(Xb, aoff1 + kb + 32);
#pragma unroll
        for (int j = 0; j < 4; j++) bB[j] = ldfrag_bf16(Wob, boff[j] + kb + 32);
#pragma unroll
        for (int j = 0; j < 4; j++) {
            acc[0][j] = WMMA_BF16(a0A, bA[j], acc[0][j]);
            acc[1][j] = WMMA_BF16(a1A, bA[j], acc[1][j]);
        }
        if (kb + 64 < D_MODEL) {
            a0A = ldfrag_bf16(Xb, aoff0 + kb + 64);
            a1A = ldfrag_bf16(Xb, aoff1 + kb + 64);
#pragma unroll
            for (int j = 0; j < 4; j++) bA[j] = ldfrag_bf16(Wob, boff[j] + kb + 64);
        }
#pragma unroll
        for (int j = 0; j < 4; j++) {
            acc[0][j] = WMMA_BF16(a0B, bB[j], acc[0][j]);
            acc[1][j] = WMMA_BF16(a1B, bB[j], acc[1][j]);
        }
    }
#pragma unroll
    for (int j = 0; j < 4; j++) {
        int n = nBase + 16 * j + ln;
        float bb = bo[n], g = gamma[n];
#pragma unroll
        for (int t = 0; t < 2; t++)
#pragma unroll
            for (int r = 0; r < 8; r++) {
                int m = mBase + 16 * t + r + 8 * half;
                out[(size_t)m * D_MODEL + n] = g * (acc[t][j][r] + bb);
            }
    }
}

extern "C" void kernel_launch(void* const* d_in, const int* in_sizes, int n_in,
                              void* d_out, int out_size, void* d_ws, size_t ws_size,
                              hipStream_t stream) {
    const float* query = (const float*)d_in[0];
    const float* Wq    = (const float*)d_in[1];
    const float* bq    = (const float*)d_in[2];
    const float* Wk    = (const float*)d_in[3];
    const float* bk    = (const float*)d_in[4];
    const float* Wv    = (const float*)d_in[5];
    const float* bv    = (const float*)d_in[6];
    const float* Wo    = (const float*)d_in[7];
    const float* bo    = (const float*)d_in[8];
    const float* gamma = (const float*)d_in[9];

    const size_t WME = (size_t)D_MODEL * D_MODEL;              // 1M elems per weight
    const size_t TE  = (size_t)BATCH * N_HEADS * SEQ * DEPTH;  // 4M elems per tensor

    unsigned short* wqb = (unsigned short*)d_ws;
    unsigned short* wkb = wqb + WME;
    unsigned short* wvb = wkb + WME;
    unsigned short* wob = wvb + WME;
    unsigned short* qws = wob + WME;
    unsigned short* kws = qws + TE;
    unsigned short* vws = kws + TE;
    unsigned short* xb  = vws + TE;
    unsigned short* ows = xb;   // alias: xb dead after kernel 1

    cvt_bf16<<<dim3(512, 8), 256, 0, stream>>>(
        query, Wq, Wk, Wv, Wo, xb, wqb, wkb, wvb, wob);

    qkv_rope_wmma<<<dim3(BATCH * SEQ / 32, N_HEADS, 3), 32, 0, stream>>>(
        xb, wqb, wkb, wvb, bq, bk, bv, qws, kws, vws);

    attn_flash_wmma<<<dim3(SEQ / 128, N_HEADS, BATCH), 256, 0, stream>>>(
        qws, kws, vws, ows);

    oproj_wmma<<<dim3(BATCH * SEQ / 32, D_MODEL / 64), 32, 0, stream>>>(
        ows, wob, bo, gamma, (float*)d_out);
}
